// ProTCL_32049045963144
// MI455X (gfx1250) — matrix operand order repaired and checked by reading, not compiled, not run on hardware
//
#include <hip/hip_runtime.h>

// ---------------------------------------------------------------------------
// ProTCL fused scorer for MI455X (gfx1250, wave32, WMMA bf16 16x16x32).
//
// logits[s,n] = W3 . relu( W2 @ relu(hp[s] + hl[n] + b1) + b2 ) + b3
//   hp = (P_f @ W_p^T) @ W1[:, :512]^T        [64, 512]
//   hl = (label_emb @ W_l^T) @ W1[:, 512:]^T  [5000, 512]
// Dominant GEMM (168 GFLOP) is fused: the [S*NL,512] intermediates never
// touch HBM. All matmuls run on v_wmma_f32_16x16x32_bf16 (fp32 accum).
// Phase-1 GEMMs stream weight tiles through LDS with
// GLOBAL_LOAD_ASYNC_TO_LDS_B128 (ASYNCcnt double-buffering).
// ---------------------------------------------------------------------------

typedef __attribute__((ext_vector_type(16))) __bf16 v16bf;
typedef __attribute__((ext_vector_type(8)))  __bf16 v8bf;
typedef __attribute__((ext_vector_type(8)))  float  v8f;

#define SEQ   64
#define NLBL  5000
#define PDIM  1024
#define LDIM  768
#define DIMH  512
#define ODIM  512

#define LE_ROWS 5008   // 313 * 16
#define HL_ROWS 5056   // 79 * 64 (phase-2 row blocks)

__device__ __forceinline__ unsigned short f2bf_bits(float f) {
  unsigned u = __builtin_bit_cast(unsigned, f);
  return (unsigned short)((u + 0x7FFFu + ((u >> 16) & 1u)) >> 16);  // RNE
}

__device__ __forceinline__ unsigned lds_addr32(const void* p) {
  // generic LDS address: low 32 bits are the LDS byte offset (ISA 10.2)
  return (unsigned)(unsigned long long)p;
}

__device__ __forceinline__ v16bf cat8(v8bf lo, v8bf hi) {
  return __builtin_shufflevector(lo, hi, 0, 1, 2, 3, 4, 5, 6, 7,
                                 8, 9, 10, 11, 12, 13, 14, 15);
}

// ---------------- fp32 -> bf16 conversion (vectorized) ----------------------
__global__ __launch_bounds__(256) void cvt_f32_bf16(const float* __restrict__ src,
                                                    unsigned short* __restrict__ dst,
                                                    int n4) {
  int i = blockIdx.x * 256 + threadIdx.x;
  int stride = gridDim.x * 256;
  for (; i < n4; i += stride) {
    float4 v = ((const float4*)src)[i];
    ushort4 o;
    o.x = f2bf_bits(v.x); o.y = f2bf_bits(v.y);
    o.z = f2bf_bits(v.z); o.w = f2bf_bits(v.w);
    ((ushort4*)dst)[i] = o;
  }
}

// Split W1 [512,1024] into bf16 W1p = W1[:, :512], W1l = W1[:, 512:].
__global__ __launch_bounds__(256) void cvt_split_w1(const float* __restrict__ W1,
                                                    unsigned short* __restrict__ W1p,
                                                    unsigned short* __restrict__ W1l) {
  int i = blockIdx.x * 256 + threadIdx.x;   // over 512*256 float4 chunks
  if (i >= 512 * 256) return;
  int row = i >> 8, c4 = i & 255;
  float4 v = ((const float4*)W1)[i];
  ushort4 o;
  o.x = f2bf_bits(v.x); o.y = f2bf_bits(v.y);
  o.z = f2bf_bits(v.z); o.w = f2bf_bits(v.w);
  if (c4 < 128) ((ushort4*)W1p)[row * 128 + c4] = o;
  else          ((ushort4*)W1l)[row * 128 + (c4 - 128)] = o;
}

// ---------------- generic C = A @ B^T, bf16 in, fp32 accum ------------------
// A [Mpad,K] bf16 row-major, B [N,K] bf16 row-major, N = 512, K % 64 == 0.
// 4 waves/block; each wave computes a 16x64 tile (4 WMMA accumulators).
// B tiles [64n x 64k] are double-buffered through LDS with async DMA:
// wave 0 issues GLOBAL_LOAD_ASYNC_TO_LDS_B128, fences with s_wait_asynccnt
// + workgroup barrier; all 4 waves (same n0) share the staged tile.
// LDS pitch = 72 elements (36 dwords) -> 16 lanes of each ds_load_b128 group
// start at 16 distinct bank quads (36t mod 64 distinct for t=0..15).
#define BPITCH 72

__device__ __forceinline__ void async_b_chunk(const __bf16* __restrict__ B,
                                              int n0, int kc, int K,
                                              unsigned ldsBase, int lane) {
#pragma unroll
  for (int j = 0; j < 16; ++j) {
    int idx = j * 32 + lane;          // 512 lane-transfers of 16B
    int row = idx >> 3, seg = idx & 7;
    unsigned lds = ldsBase + row * (BPITCH * 2) + seg * 16;
    unsigned long long ga = (unsigned long long)(const void*)
        (B + (size_t)(n0 + row) * K + kc + seg * 8);
    asm volatile("global_load_async_to_lds_b128 %0, %1, off"
                 :: "v"(lds), "v"(ga) : "memory");
  }
}

template <bool OUT_BF16>
__global__ __launch_bounds__(128) void gemm_bf16_nt(const __bf16* __restrict__ A,
                                                    const __bf16* __restrict__ B,
                                                    float* __restrict__ Cf,
                                                    unsigned short* __restrict__ Cb,
                                                    int Mtiles, int K, int ldc) {
  __shared__ __align__(16) unsigned short bS[2][64 * BPITCH];  // 18 KB
  const int lane = threadIdx.x & 31;
  const int wave = threadIdx.x >> 5;
  const int rt = blockIdx.x * 4 + wave;
  const bool active = rt < Mtiles;
  const int rtc = active ? rt : (Mtiles - 1);
  const int g = lane >> 4, t = lane & 15;
  const int m0 = rtc * 16;
  const int n0 = blockIdx.y * 64;

  const __bf16* __restrict__ ar = A + (size_t)(m0 + t) * K;
  const unsigned bBase0 = lds_addr32(&bS[0][0]);
  const unsigned bBase1 = lds_addr32(&bS[1][0]);

  if (wave == 0) async_b_chunk(B, n0, 0, K, bBase0, lane);

  v8f acc[4] = {};
  int buf = 0;
  for (int kc = 0; kc < K; kc += 64) {
    if (wave == 0) asm volatile("s_wait_asynccnt 0" ::: "memory");
    __syncthreads();  // staged chunk visible to all waves
    if (wave == 0 && kc + 64 < K)
      async_b_chunk(B, n0, kc + 64, K, buf ? bBase0 : bBase1, lane);

#pragma unroll
    for (int kk = 0; kk < 64; kk += 32) {
      v8bf alo = *(const v8bf*)(ar + kc + kk + 16 * g);
      v8bf ahi = *(const v8bf*)(ar + kc + kk + 16 * g + 8);
      v16bf a = cat8(alo, ahi);
#pragma unroll
      for (int c = 0; c < 4; ++c) {
        const unsigned short* bp =
            &bS[buf][(c * 16 + t) * BPITCH + kk + 16 * g];
        v16bf b = cat8(*(const v8bf*)bp, *(const v8bf*)(bp + 8));
        acc[c] = __builtin_amdgcn_wmma_f32_16x16x32_bf16(
            false, a, false, b, (short)0, acc[c], false, false);
      }
    }
    buf ^= 1;
  }
  if (!active) return;
  // C/D layout: VGPR i -> row m0 + i + 8g, col n0 + c*16 + t
#pragma unroll
  for (int c = 0; c < 4; ++c) {
    const int n = n0 + c * 16 + t;
#pragma unroll
    for (int i = 0; i < 8; ++i) {
      const size_t m = (size_t)(m0 + i + 8 * g);
      if (OUT_BF16) Cb[m * ldc + n] = f2bf_bits(acc[c][i]);
      else          Cf[m * ldc + n] = acc[c][i];
    }
  }
}

// ---------------- fused tail: build-h -> GEMM(W2) -> relu -> dot(W3) --------
// Block: 128 threads (4 waves), one s and a 64-row block of n.
// LDS: 64x512 bf16 activation tile (64KB), XOR-swizzled (16B grain) so the
// 16-lane ds_load_b128 fragment reads are conflict-free; after compute the
// same LDS is reused as the 128-float logit reduction scratch.
// Wave w owns rows (w&1)*32..+31 for o-chunks (w>>1)*4..+3: per k-step it
// issues 2 A-frags + 4 B-frags for 8 WMMAs (1.5 loads/WMMA).
__global__ __launch_bounds__(128) void fused_tail(const float* __restrict__ hp,
                                                  const float* __restrict__ hl,
                                                  const float* __restrict__ b1,
                                                  const __bf16* __restrict__ W2b,
                                                  const float* __restrict__ b2,
                                                  const float* __restrict__ W3,
                                                  const float* __restrict__ b3,
                                                  float* __restrict__ out) {
  union SmemU { unsigned short a[64 * 512]; float red[128]; };
  __shared__ __align__(16) SmemU sm;  // 64 KB
  const int s  = blockIdx.x;
  const int n0 = blockIdx.y * 64;
  const int tid = threadIdx.x;

  {  // build a-tile: row r holds relu(hp[s] + hl[n0+r] + b1) as bf16
    const int r  = tid >> 1;
    const int c0 = (tid & 1) * 256;
    const float* __restrict__ hpr = hp + s * DIMH + c0;
    const float* __restrict__ hlr = hl + (size_t)(n0 + r) * DIMH + c0;
    const float* __restrict__ b1r = b1 + c0;
    char* rowp = (char*)sm.a + r * 1024;
    const int swz = (r & 15) << 4;
#pragma unroll 4
    for (int j = 0; j < 256; j += 4) {
      float4 p = *(const float4*)(hpr + j);
      float4 l = *(const float4*)(hlr + j);
      float4 bb = *(const float4*)(b1r + j);
      ushort4 o;
      o.x = f2bf_bits(fmaxf(p.x + l.x + bb.x, 0.f));
      o.y = f2bf_bits(fmaxf(p.y + l.y + bb.y, 0.f));
      o.z = f2bf_bits(fmaxf(p.z + l.z + bb.z, 0.f));
      o.w = f2bf_bits(fmaxf(p.w + l.w + bb.w, 0.f));
      *(ushort4*)(rowp + (((c0 + j) * 2) ^ swz)) = o;
    }
  }
  __syncthreads();

  const int lane = tid & 31, wave = tid >> 5;
  const int g = lane >> 4, t = lane & 15;
  const int rows0 = (wave & 1) * 32;   // this wave's 32 a-tile rows
  const int oh    = wave >> 1;         // this wave's half of the o-chunks
  const char* __restrict__ rowpA = (const char*)sm.a + (rows0 + t) * 1024;
  const char* __restrict__ rowpB = rowpA + 16 * 1024;
  const int swz = t << 4;              // (rows0{,+16} + t) & 15 == t

  float logit[2][8] = {};
#pragma unroll 1
  for (int ob = oh * 4; ob < oh * 4 + 4; ++ob) {  // 4 chunks of 64 outputs
    const int o0 = ob * 64;
    v8f acc[2][4] = {};
#pragma unroll 2
    for (int k0 = 0; k0 < DIMH; k0 += 32) {
      const int off1 = (k0 * 2 + 32 * g) ^ swz;
      const int off2 = (k0 * 2 + 32 * g + 16) ^ swz;
      v16bf a0 = cat8(*(const v8bf*)(rowpA + off1), *(const v8bf*)(rowpA + off2));
      v16bf a1 = cat8(*(const v8bf*)(rowpB + off1), *(const v8bf*)(rowpB + off2));
#pragma unroll
      for (int c = 0; c < 4; ++c) {
        v16bf b = *(const v16bf*)(W2b + (size_t)(o0 + c * 16 + t) * DIMH +
                                  k0 + 16 * g);
        acc[0][c] = __builtin_amdgcn_wmma_f32_16x16x32_bf16(
            false, a0, false, b, (short)0, acc[0][c], false, false);
        acc[1][c] = __builtin_amdgcn_wmma_f32_16x16x32_bf16(
            false, a1, false, b, (short)0, acc[1][c], false, false);
      }
    }
    // epilogue: relu(+b2) then scale by W3, fold into per-row partials
#pragma unroll
    for (int c = 0; c < 4; ++c) {
      const int n = o0 + c * 16 + t;
      const float bb = b2[n], w = W3[n];
#pragma unroll
      for (int i = 0; i < 8; ++i) {
        logit[0][i] += fmaxf(acc[0][c][i] + bb, 0.f) * w;
        logit[1][i] += fmaxf(acc[1][c][i] + bb, 0.f) * w;
      }
    }
  }
  // reduce over the 16 lanes of each half-wave (N dim of the tile)
#pragma unroll
  for (int m = 1; m < 16; m <<= 1)
#pragma unroll
    for (int i = 0; i < 8; ++i) {
      logit[0][i] += __shfl_xor(logit[0][i], m, 32);
      logit[1][i] += __shfl_xor(logit[1][i], m, 32);
    }

  __syncthreads();  // done reading sm.a -> reuse as reduction scratch
  if (t == 0) {
#pragma unroll
    for (int i = 0; i < 8; ++i) {
      sm.red[oh * 64 + rows0 + 8 * g + i]      = logit[0][i];
      sm.red[oh * 64 + rows0 + 16 + 8 * g + i] = logit[1][i];
    }
  }
  __syncthreads();
  if (tid < 64) {
    const int n = n0 + tid;
    if (n < NLBL)
      out[s * NLBL + n] = sm.red[tid] + sm.red[64 + tid] + b3[0];
  }
}

// ---------------------------------------------------------------------------
extern "C" void kernel_launch(void* const* d_in, const int* in_sizes, int n_in,
                              void* d_out, int out_size, void* d_ws, size_t ws_size,
                              hipStream_t stream) {
  (void)in_sizes; (void)n_in; (void)out_size; (void)ws_size;
  const float* P_f       = (const float*)d_in[0];
  const float* label_emb = (const float*)d_in[1];
  const float* W_p       = (const float*)d_in[2];
  const float* W_l       = (const float*)d_in[3];
  const float* W1        = (const float*)d_in[4];
  const float* b1        = (const float*)d_in[5];
  const float* W2        = (const float*)d_in[6];
  const float* b2        = (const float*)d_in[7];
  const float* W3        = (const float*)d_in[8];
  const float* b3        = (const float*)d_in[9];
  float* out = (float*)d_out;

  // workspace carve-out (~27 MB)
  char* ws = (char*)d_ws;
  size_t off = 0;
  auto alloc = [&](size_t bytes) -> void* {
    off = (off + 255) & ~(size_t)255;
    void* p = ws + off;
    off += bytes;
    return p;
  };
  unsigned short* Pf_bf   = (unsigned short*)alloc((size_t)SEQ * PDIM * 2);
  unsigned short* Lemb_bf = (unsigned short*)alloc((size_t)LE_ROWS * LDIM * 2);
  unsigned short* Wp_bf   = (unsigned short*)alloc((size_t)DIMH * PDIM * 2);
  unsigned short* Wl_bf   = (unsigned short*)alloc((size_t)DIMH * LDIM * 2);
  unsigned short* W1p_bf  = (unsigned short*)alloc((size_t)ODIM * DIMH * 2);
  unsigned short* W1l_bf  = (unsigned short*)alloc((size_t)ODIM * DIMH * 2);
  unsigned short* W2_bf   = (unsigned short*)alloc((size_t)ODIM * ODIM * 2);
  unsigned short* Pe_bf   = (unsigned short*)alloc((size_t)SEQ * DIMH * 2);
  unsigned short* Le_bf   = (unsigned short*)alloc((size_t)LE_ROWS * DIMH * 2);
  float*          hp_f    = (float*)alloc((size_t)SEQ * DIMH * 4);
  float*          hl_f    = (float*)alloc((size_t)HL_ROWS * DIMH * 4);

  // 1) convert activations + weights to bf16
  auto cvt = [&](const float* src, unsigned short* dst, int n) {
    int n4 = n / 4;
    cvt_f32_bf16<<<(n4 + 255) / 256, 256, 0, stream>>>(src, dst, n4);
  };
  cvt(P_f, Pf_bf, SEQ * PDIM);
  cvt(label_emb, Lemb_bf, NLBL * LDIM);
  cvt(W_p, Wp_bf, DIMH * PDIM);
  cvt(W_l, Wl_bf, DIMH * LDIM);
  cvt(W2, W2_bf, ODIM * ODIM);
  cvt_split_w1<<<(512 * 256 + 255) / 256, 256, 0, stream>>>(W1, W1p_bf, W1l_bf);

  // 2) projections: Pe = Pf @ Wp^T, Le = Lemb @ Wl^T  (bf16 out)
  gemm_bf16_nt<true><<<dim3(1, 8), 128, 0, stream>>>(
      (const __bf16*)Pf_bf, (const __bf16*)Wp_bf, nullptr, Pe_bf,
      SEQ / 16, PDIM, DIMH);
  gemm_bf16_nt<true><<<dim3(79, 8), 128, 0, stream>>>(
      (const __bf16*)Lemb_bf, (const __bf16*)Wl_bf, nullptr, Le_bf,
      LE_ROWS / 16, LDIM, DIMH);

  // 3) hidden halves: hp = Pe @ W1p^T, hl = Le @ W1l^T  (fp32 out)
  gemm_bf16_nt<false><<<dim3(1, 8), 128, 0, stream>>>(
      (const __bf16*)Pe_bf, (const __bf16*)W1p_bf, hp_f, nullptr,
      SEQ / 16, DIMH, ODIM);
  gemm_bf16_nt<false><<<dim3(79, 8), 128, 0, stream>>>(
      (const __bf16*)Le_bf, (const __bf16*)W1l_bf, hl_f, nullptr,
      LE_ROWS / 16, DIMH, ODIM);

  // 4) fused dominant stage: 64 s-blocks x 79 n-blocks of 64 rows
  fused_tail<<<dim3(SEQ, HL_ROWS / 64), 128, 0, stream>>>(
      hp_f, hl_f, b1, (const __bf16*)W2_bf, b2, W3, b3, out);
}
